// MultiHeadAttention_26998164423233
// MI455X (gfx1250) — matrix operand order repaired and checked by reading, not compiled, not run on hardware
//
#include <hip/hip_runtime.h>
#include <hip/hip_bf16.h>

// ---------------------------------------------------------------------------
// MultiHeadAttention (B=4, T=2048, D=512, H=8, K=64) via CDNA5 WMMA bf16
// + Tensor Data Mover (TDM) double-buffered B-panel staging in the GEMMs.
// ---------------------------------------------------------------------------

#define B_BATCH 4
#define T_SEQ   2048
#define D_MODEL 512
#define N_HEADS 8
#define HEAD_K  64
#define M_ROWS  (B_BATCH * T_SEQ)   // 8192

typedef __attribute__((ext_vector_type(16))) __bf16 v16bf;
typedef __attribute__((ext_vector_type(8)))  __bf16 v8bf;
typedef __attribute__((ext_vector_type(8)))  float  v8f;
typedef __attribute__((ext_vector_type(4)))  unsigned int u32x4;
typedef __attribute__((ext_vector_type(8)))  int i32x8;
typedef __attribute__((ext_vector_type(4)))  int i32x4;

#define CAT16(lo, hi) __builtin_shufflevector((lo), (hi), 0,1,2,3,4,5,6,7,8,9,10,11,12,13,14,15)

__device__ __forceinline__ v8f wmma_bf16(v16bf a, v16bf b, v8f c) {
    return __builtin_amdgcn_wmma_f32_16x16x32_bf16(false, a, false, b, (short)0, c, false, false);
}

// ---- TDM: DMA a 32(K-rows) x 64(N-cols) bf16 panel from a [512,512] bf16
// matrix (row stride 512 elements) into LDS at lds_off.  2D descriptor:
// data_size=2B, tile_dim0=64, tile_dim1=32, tensor_dim0_stride=512, type=2.
__device__ __forceinline__ void tdm_load_B_panel(const __bf16* gaddr, unsigned lds_off) {
    unsigned long long ga = (unsigned long long)(size_t)gaddr;
    unsigned ga_lo = (unsigned)__builtin_amdgcn_readfirstlane((int)(unsigned)ga);
    unsigned ga_hi = (unsigned)__builtin_amdgcn_readfirstlane((int)(unsigned)(ga >> 32));
    unsigned ldso  = (unsigned)__builtin_amdgcn_readfirstlane((int)lds_off);

    u32x4 g0;
    g0[0] = 1u;                                   // count=1 user descriptor
    g0[1] = ldso;                                 // lds_addr
    g0[2] = ga_lo;                                // global_addr[31:0]
    g0[3] = (ga_hi & 0x1FFFFFFu) | (2u << 30);    // global_addr[56:32] | type=2

    i32x8 g1;
    g1[0] = 0x00010000;                           // data_size=1 (2 bytes)
    g1[1] = (int)(512u << 16);                    // tensor_dim0 = 512
    g1[2] = (int)(512u << 16);                    // tensor_dim1 = 512
    g1[3] = (int)(64u  << 16);                    // tile_dim0 = 64
    g1[4] = 32;                                   // tile_dim1 = 32 (tile_dim2 = 0)
    g1[5] = 512;                                  // tensor_dim0_stride = 512
    g1[6] = 0;
    g1[7] = 0;

    i32x4 g2 = (i32x4)0, g3 = (i32x4)0;
#if defined(__clang_major__) && (__clang_major__ >= 23)
    i32x8 g4 = (i32x8)0;
    __builtin_amdgcn_tensor_load_to_lds(g0, g1, g2, g3, g4, 0);
#else
    __builtin_amdgcn_tensor_load_to_lds(g0, g1, g2, g3, 0);
#endif
}

// ---- prep kernels ----------------------------------------------------------

__global__ void k_f32_to_bf16(const float* __restrict__ in, __bf16* __restrict__ out, int n) {
    int i = blockIdx.x * blockDim.x + threadIdx.x;
    if (i < n) out[i] = (__bf16)in[i];
}

// 512x512 transpose: out[d*512 + n] = (bf16) in[n*512 + d]
__global__ void k_transpose512(const float* __restrict__ in, __bf16* __restrict__ out) {
    int idx = blockIdx.x * blockDim.x + threadIdx.x;   // over 512*512
    int n = idx >> 9;
    int d = idx & 511;
    out[d * 512 + n] = (__bf16)in[n * 512 + d];
}

// Kb [B*T, 512] (col = h*64+d)  ->  Kt [B,H,64,T]
__global__ void k_build_kt(const __bf16* __restrict__ Kb, __bf16* __restrict__ Kt) {
    int idx = blockIdx.x * blockDim.x + threadIdx.x;   // over B*T*512
    int col = idx & 511;
    int bt  = idx >> 9;
    int b   = bt >> 11;
    int t   = bt & (T_SEQ - 1);
    int h   = col >> 6;
    int d   = col & 63;
    Kt[(((size_t)(b * N_HEADS + h) * HEAD_K + d) * T_SEQ) + t] = Kb[idx];
}

// ---- WMMA GEMM: C[M,512] = A[M,512] * Bt[512,512] --------------------------
// One wave per 32x64 output tile.  B panel staged in LDS by TDM, double
// buffered; A read directly from global (16B loads, A-layout).
// BIAS=false: bf16 out.  BIAS=true: f32 out + bias.

template <bool BIAS>
__global__ __launch_bounds__(32)
void k_gemm(const __bf16* __restrict__ A, const __bf16* __restrict__ Bt,
            __bf16* __restrict__ Cb, float* __restrict__ Cf,
            const float* __restrict__ bias) {
    __shared__ __align__(32) __bf16 Bpanel[2][32][64];   // 2 x 4KB

    const int lane = threadIdx.x;
    const int m0 = blockIdx.x * 32;
    const int n0 = blockIdx.y * 64;
    const int K  = D_MODEL;
    const int N  = D_MODEL;
    const int arow  = m0 + (lane & 15);
    const int klane = (lane < 16) ? 0 : 8;

    // prologue: DMA first B panel
    tdm_load_B_panel(Bt + (size_t)0 * N + n0, (unsigned)(size_t)&Bpanel[0][0][0]);

    v8f acc[2][4];
#pragma unroll
    for (int mt = 0; mt < 2; ++mt)
#pragma unroll
        for (int nt = 0; nt < 4; ++nt) acc[mt][nt] = v8f{};

    int buf = 0;
    for (int kk = 0; kk < K; kk += 32) {
        __builtin_amdgcn_s_wait_tensorcnt(0);
        __asm__ volatile("" ::: "memory");
        if (kk + 32 < K)
            tdm_load_B_panel(Bt + (size_t)(kk + 32) * N + n0,
                             (unsigned)(size_t)&Bpanel[buf ^ 1][0][0]);

        // A tiles for the two 16-row strips
        v16bf a[2];
#pragma unroll
        for (int mt = 0; mt < 2; ++mt) {
            const __bf16* ap = A + (size_t)(arow + mt * 16) * K + kk + klane;
            v8bf lo = *(const v8bf*)(ap);
            v8bf hi = *(const v8bf*)(ap + 16);
            a[mt] = CAT16(lo, hi);
        }

#pragma unroll
        for (int nt = 0; nt < 4; ++nt) {
            v16bf b;
#pragma unroll
            for (int s = 0; s < 16; ++s)
                b[s] = Bpanel[buf][(s < 8) ? (klane + s) : (8 + klane + s)][nt * 16 + (lane & 15)];
            acc[0][nt] = wmma_bf16(a[0], b, acc[0][nt]);
            acc[1][nt] = wmma_bf16(a[1], b, acc[1][nt]);
        }
        // ensure LDS reads of this panel retired before the TDM two steps out
        __asm__ volatile("s_wait_dscnt 0" ::: "memory");
        buf ^= 1;
    }

    const int colBase = n0 + (lane & 15);
    const int rowBase = m0 + ((lane < 16) ? 0 : 8);
#pragma unroll
    for (int mt = 0; mt < 2; ++mt) {
#pragma unroll
        for (int nt = 0; nt < 4; ++nt) {
            if constexpr (BIAS) {
                float bb = bias[colBase + nt * 16];
#pragma unroll
                for (int r = 0; r < 8; ++r)
                    Cf[(size_t)(rowBase + mt * 16 + r) * N + colBase + nt * 16] =
                        acc[mt][nt][r] + bb;
            } else {
#pragma unroll
                for (int r = 0; r < 8; ++r)
                    Cb[(size_t)(rowBase + mt * 16 + r) * N + colBase + nt * 16] =
                        (__bf16)acc[mt][nt][r];
            }
        }
    }
}

// ---- causal flash attention: one wave per (b, h, 16 query rows) ------------

__global__ __launch_bounds__(32)
void k_attn(const __bf16* __restrict__ Qb,   // [B*T, 512] (col = h*64+k)
            const __bf16* __restrict__ Kt,   // [B,H,64,T]
            const __bf16* __restrict__ Vb,   // [B*T, 512]
            __bf16* __restrict__ Ob) {       // [B*T, 512]
    __shared__ __align__(32) __bf16 Plds[16][32];

    const int lane = threadIdx.x;
    const int t0 = blockIdx.x * 16;
    const int h  = blockIdx.y;
    const int b  = blockIdx.z;
    const float scale = 0.04419417382415922f;   // 512^-0.5 (ref scales by d_model)

    const int klane   = (lane < 16) ? 0 : 8;
    const int rowbase = (lane < 16) ? 0 : 8;
    const int ncol    = lane & 15;

    v16bf aQ[2];
    {
        const size_t qrow = (size_t)(b * T_SEQ + t0 + (lane & 15));
#pragma unroll
        for (int ch = 0; ch < 2; ++ch) {
            const __bf16* qp = Qb + qrow * D_MODEL + h * HEAD_K + ch * 32 + klane;
            v8bf lo = *(const v8bf*)(qp);
            v8bf hi = *(const v8bf*)(qp + 16);
            aQ[ch] = CAT16(lo, hi);
        }
    }

    v8f acc[4] = {v8f{}, v8f{}, v8f{}, v8f{}};
    float mrow[8], lrow[8];
#pragma unroll
    for (int r = 0; r < 8; ++r) { mrow[r] = -1e30f; lrow[r] = 0.0f; }

    const size_t ktBase = ((size_t)(b * N_HEADS + h) * HEAD_K) * T_SEQ;

    for (int s0 = 0; s0 < t0 + 16; s0 += 32) {
        v8f st[2];
#pragma unroll
        for (int j = 0; j < 2; ++j) {
            v8f c = v8f{};
#pragma unroll
            for (int ch = 0; ch < 2; ++ch) {
                v16bf bK;
#pragma unroll
                for (int s = 0; s < 16; ++s)
                    bK[s] = Kt[ktBase + (size_t)(ch * 32 + ((s < 8) ? (klane + s) : (8 + klane + s))) * T_SEQ + s0 + j * 16 + ncol];
                c = wmma_bf16(aQ[ch], bK, c);
            }
            st[j] = c;
        }

#pragma unroll
        for (int r = 0; r < 8; ++r) {
            const int trow = t0 + rowbase + r;
            float v0 = st[0][r] * scale;
            float v1 = st[1][r] * scale;
            if (s0 + ncol      > trow) v0 = -1e30f;
            if (s0 + 16 + ncol > trow) v1 = -1e30f;

            float mx = fmaxf(v0, v1);
#pragma unroll
            for (int off = 1; off < 16; off <<= 1)
                mx = fmaxf(mx, __shfl_xor(mx, off, 32));

            const float mnew  = fmaxf(mrow[r], mx);
            const float alpha = __expf(mrow[r] - mnew);
            mrow[r] = mnew;

            const float p0 = __expf(v0 - mnew);
            const float p1 = __expf(v1 - mnew);
            float ps = p0 + p1;
#pragma unroll
            for (int off = 1; off < 16; off <<= 1)
                ps += __shfl_xor(ps, off, 32);

            lrow[r] = lrow[r] * alpha + ps;
#pragma unroll
            for (int kt = 0; kt < 4; ++kt) acc[kt][r] *= alpha;

            Plds[rowbase + r][ncol]      = (__bf16)p0;
            Plds[rowbase + r][16 + ncol] = (__bf16)p1;
        }

        v16bf aP;
        {
            v8bf lo = *(const v8bf*)(&Plds[lane & 15][klane]);
            v8bf hi = *(const v8bf*)(&Plds[lane & 15][klane + 16]);
            aP = CAT16(lo, hi);
        }

        const size_t vrow = (size_t)(b * T_SEQ + s0);
#pragma unroll
        for (int kt = 0; kt < 4; ++kt) {
            v16bf bV;
#pragma unroll
            for (int s = 0; s < 16; ++s)
                bV[s] = Vb[(vrow + ((s < 8) ? (klane + s) : (8 + klane + s))) * D_MODEL + h * HEAD_K + kt * 16 + ncol];
            acc[kt] = wmma_bf16(aP, bV, acc[kt]);
        }
    }

    const size_t obase = (size_t)(b * T_SEQ + t0 + rowbase) * D_MODEL + h * HEAD_K;
#pragma unroll
    for (int kt = 0; kt < 4; ++kt)
#pragma unroll
        for (int r = 0; r < 8; ++r)
            Ob[obase + (size_t)r * D_MODEL + kt * 16 + ncol] = (__bf16)(acc[kt][r] / lrow[r]);
}

// ---------------------------------------------------------------------------

extern "C" void kernel_launch(void* const* d_in, const int* in_sizes, int n_in,
                              void* d_out, int out_size, void* d_ws, size_t ws_size,
                              hipStream_t stream) {
    const float* x_q = (const float*)d_in[0];
    const float* Wq  = (const float*)d_in[1];
    const float* Wk  = (const float*)d_in[2];
    const float* Wv  = (const float*)d_in[3];
    const float* Wp  = (const float*)d_in[4];
    const float* bp  = (const float*)d_in[5];
    float* out = (float*)d_out;

    char* ws = (char*)d_ws;
    const size_t SZ_BT512 = (size_t)M_ROWS * D_MODEL * sizeof(__bf16);   // 8 MiB
    const size_t SZ_W     = (size_t)D_MODEL * D_MODEL * sizeof(__bf16);  // 512 KiB
    __bf16* Xb  = (__bf16*)(ws);
    __bf16* Qb  = (__bf16*)(ws + SZ_BT512);
    __bf16* Kb  = (__bf16*)(ws + 2 * SZ_BT512);
    __bf16* Vb  = (__bf16*)(ws + 3 * SZ_BT512);
    __bf16* Ob  = (__bf16*)(ws + 4 * SZ_BT512);
    __bf16* Kt  = (__bf16*)(ws + 5 * SZ_BT512);
    __bf16* Wqt = (__bf16*)(ws + 6 * SZ_BT512);
    __bf16* Wkt = (__bf16*)(ws + 6 * SZ_BT512 + SZ_W);
    __bf16* Wvt = (__bf16*)(ws + 6 * SZ_BT512 + 2 * SZ_W);
    __bf16* Wpt = (__bf16*)(ws + 6 * SZ_BT512 + 3 * SZ_W);

    const int nX = M_ROWS * D_MODEL;

    k_f32_to_bf16<<<(nX + 255) / 256, 256, 0, stream>>>(x_q, Xb, nX);

    const int nW = D_MODEL * D_MODEL;
    k_transpose512<<<(nW + 255) / 256, 256, 0, stream>>>(Wq, Wqt);
    k_transpose512<<<(nW + 255) / 256, 256, 0, stream>>>(Wk, Wkt);
    k_transpose512<<<(nW + 255) / 256, 256, 0, stream>>>(Wv, Wvt);
    k_transpose512<<<(nW + 255) / 256, 256, 0, stream>>>(Wp, Wpt);

    dim3 gGemm(M_ROWS / 32, D_MODEL / 64);
    k_gemm<false><<<gGemm, 32, 0, stream>>>(Xb, Wqt, Qb, nullptr, nullptr);
    k_gemm<false><<<gGemm, 32, 0, stream>>>(Xb, Wkt, Kb, nullptr, nullptr);
    k_gemm<false><<<gGemm, 32, 0, stream>>>(Xb, Wvt, Vb, nullptr, nullptr);

    k_build_kt<<<(nX + 255) / 256, 256, 0, stream>>>(Kb, Kt);

    dim3 gAttn(T_SEQ / 16, N_HEADS, B_BATCH);
    k_attn<<<gAttn, 32, 0, stream>>>(Qb, Kt, Vb, Ob);

    k_gemm<true><<<gGemm, 32, 0, stream>>>(Ob, Wpt, nullptr, out, bp);
}
